// DeformConv_40905268527556
// MI455X (gfx1250) — matrix operand order repaired and checked by reading, not compiled, not run on hardware
//
#include <hip/hip_runtime.h>
#include <hip/hip_bf16.h>

typedef __attribute__((ext_vector_type(16))) _Float16 v16h;
typedef __attribute__((ext_vector_type(8)))  float    v8f;

// Problem constants (from reference setup_inputs)
#define NN    2
#define CIN   256
#define COUT  256
#define HH    64
#define WW    64
#define KK    9                     // 3x3
#define KDIM  (CIN*KK)              // 2304
#define KSTEPS (KDIM/32)            // 72 wmma k-steps total
#define SPW   32                    // spatial positions per workgroup (2 B tiles)
#define CPASS_CH 64                 // channels per LDS pass
#define NPASS (CIN/CPASS_CH)        // 4
#define KDIM_PASS (CPASS_CH*KK)     // 576
#define KSTEPS_PASS (KDIM_PASS/32)  // 18

// ---------------------------------------------------------------------------
// Kernel 0: convert f32 weight [Cout, Cin*9] into f16 WMMA A-fragments.
// Fragment layout: frag[((mt*72 + kt)*32 + lane)*16 + e]
//   lane<16 : M=lane,    e=0..15 -> K=e      (K local to step)
//   lane>=16: M=lane-16, e=0..15 -> K=e+16
// => Kloc = e + (lane & 16)
// ---------------------------------------------------------------------------
__global__ void pack_weight_f16(const float* __restrict__ w,
                                _Float16* __restrict__ frag) {
    int g = blockIdx.x * 256 + threadIdx.x;      // exactly 16*72*32*16 = 589824
    int e    = g & 15;
    int lane = (g >> 4) & 31;
    int t    = g >> 9;            // mt*72 + kt
    int kt   = t % KSTEPS;
    int mt   = t / KSTEPS;
    int Kloc = e + (lane & 16);
    int M    = lane & 15;
    int kdim = kt * 32 + Kloc;
    int cout = mt * 16 + M;
    frag[g] = (_Float16)w[cout * KDIM + kdim];
}

// ---------------------------------------------------------------------------
// Kernel 1: fused deformable im2col (bilinear, fp32) + WMMA GEMM (f16 -> f32).
// One workgroup = 32 consecutive spatial outputs x ALL 256 Cout.
// 256 threads = 8 wave32s; wave w owns a 2x2 tile grid:
//   Cout tiles {2w, 2w+1} x spatial tiles {cols 0-15, cols 16-31}
// so each A fragment and each B fragment feeds TWO wmmas (register tiling).
//
// LDS holds cols in *B-fragment-per-lane* layout so phase 2 reads each
// fragment with a single contiguous 32-byte v16h load (2x ds_load_b128,
// no register repacking):
//   half index = ktl*1024 + tile*512 + lane*16 + slot
// where for kdim-local kl (Klo = kl&31): lane = (s&15) + (Klo&16),
// slot = Klo&15, tile = s>>4, ktl = kl>>5.
// Four channel passes (64 ch each) keep LDS at 36KB.
// ---------------------------------------------------------------------------
__global__ void __launch_bounds__(256)
deform_wmma(const float* __restrict__ x,
            const float* __restrict__ offset,
            const _Float16* __restrict__ wfrag,
            float* __restrict__ out) {
    __shared__ __align__(16) _Float16 cols[KSTEPS_PASS * 2 * 32 * 16]; // 36864 B

    const int tid  = threadIdx.x;
    const int wg   = blockIdx.x;          // 0..255
    const int n    = wg >> 7;             // batch
    const int spos = (wg & 127) << 5;     // flat spatial tile base (32 wide)
    const int ho   = spos >> 6;
    const int wo0  = spos & 63;

    const int s    = tid & 31;            // spatial position within tile
    const int cg   = tid >> 5;            // channel group 0..7 (phase 1)
    const int wave = tid >> 5;            // 0..7 (wave32)
    const int lane = tid & 31;
    const int wo   = wo0 + s;

    v8f acc00 = {}, acc01 = {};           // Cout tile 0 x {spatial 0, spatial 1}
    v8f acc10 = {}, acc11 = {};           // Cout tile 1 x {spatial 0, spatial 1}

    // phase-1 constant part of the LDS half-index for this thread's position
    const int tileBase = (s >> 4) * 512 + (s & 15) * 16;

    for (int cpass = 0; cpass < NPASS; ++cpass) {
        __syncthreads();   // previous pass's LDS reads complete before overwrite

        // ---------------- Phase 1: bilinear im2col into LDS ----------------
        // Thread (cg,s): 8 channels x 9 taps, one spatial position.
        {
            const int cbase = cpass * CPASS_CH + cg * 8;
            const float* xp = x + ((long)(n * CIN + cbase)) * (HH * WW);
            for (int k = 0; k < KK; ++k) {
                const int ky = k / 3;
                const int kx = k - ky * 3;
                const float offy = offset[((n * 18 + 2 * k)     * HH + ho) * WW + wo];
                const float offx = offset[((n * 18 + 2 * k + 1) * HH + ho) * WW + wo];
                const float py = (float)(ho - 1 + ky) + offy;
                const float px = (float)(wo - 1 + kx) + offx;
                const float y0f = floorf(py);
                const float x0f = floorf(px);
                const int   y0  = (int)y0f;
                const int   x0  = (int)x0f;
                const float ly = py - y0f, lx = px - x0f;
                const float hy = 1.0f - ly, hx = 1.0f - lx;
                const float vy0 = (y0 >= 0 && y0 < HH)         ? 1.0f : 0.0f;
                const float vy1 = (y0 + 1 >= 0 && y0 + 1 < HH) ? 1.0f : 0.0f;
                const float vx0 = (x0 >= 0 && x0 < WW)         ? 1.0f : 0.0f;
                const float vx1 = (x0 + 1 >= 0 && x0 + 1 < WW) ? 1.0f : 0.0f;
                const float w00 = hy * hx * vy0 * vx0;
                const float w01 = hy * lx * vy0 * vx1;
                const float w10 = ly * hx * vy1 * vx0;
                const float w11 = ly * lx * vy1 * vx1;
                const int yc0 = min(max(y0, 0), HH - 1);
                const int yc1 = min(max(y0 + 1, 0), HH - 1);
                const int xc0 = min(max(x0, 0), WW - 1);
                const int xc1 = min(max(x0 + 1, 0), WW - 1);
                const int a00 = yc0 * WW + xc0, a01 = yc0 * WW + xc1;
                const int a10 = yc1 * WW + xc0, a11 = yc1 * WW + xc1;
#pragma unroll
                for (int j = 0; j < 8; ++j) {
                    const float* xc = xp + j * (HH * WW);
                    const float v = w00 * xc[a00] + w01 * xc[a01] +
                                    w10 * xc[a10] + w11 * xc[a11];
                    const int kl  = (cg * 8 + j) * KK + k;   // kdim local to pass
                    const int Klo = kl & 31;
                    const int idx = (kl >> 5) * 1024 + tileBase +
                                    (Klo & 16) * 16 + (Klo & 15);
                    cols[idx] = (_Float16)v;
                }
            }
        }

        __syncthreads();

        // ---------------- Phase 2: WMMA GEMM over this pass's K range ------
        {
            const int mt0 = wave * 2;
            const int mt1 = wave * 2 + 1;
            const _Float16* bbase = cols + lane * 16;   // this lane's fragments
            for (int ktl = 0; ktl < KSTEPS_PASS; ++ktl) {
                const int kt = cpass * KSTEPS_PASS + ktl;
                const _Float16* a0p = wfrag + (((mt0 * KSTEPS + kt) * 32 + lane) << 4);
                const _Float16* a1p = wfrag + (((mt1 * KSTEPS + kt) * 32 + lane) << 4);
                // prefetch next step's A fragments into cache while wmmas run
                __builtin_prefetch(a0p + 32 * 16, 0, 1);
                __builtin_prefetch(a1p + 32 * 16, 0, 1);
                const v16h a0 = *(const v16h*)a0p;
                const v16h a1 = *(const v16h*)a1p;
                // contiguous 32B fragment per lane: two ds_load_b128 each
                const v16h b0 = *(const v16h*)(bbase + ktl * 1024);
                const v16h b1 = *(const v16h*)(bbase + ktl * 1024 + 512);
                acc00 = __builtin_amdgcn_wmma_f32_16x16x32_f16(
                        false, a0, false, b0, (short)0, acc00, false, false);
                acc01 = __builtin_amdgcn_wmma_f32_16x16x32_f16(
                        false, a0, false, b1, (short)0, acc01, false, false);
                acc10 = __builtin_amdgcn_wmma_f32_16x16x32_f16(
                        false, a1, false, b0, (short)0, acc10, false, false);
                acc11 = __builtin_amdgcn_wmma_f32_16x16x32_f16(
                        false, a1, false, b1, (short)0, acc11, false, false);
            }
        }
    }

    // ---------------- Epilogue: C/D layout -> out[n][cout][ho][wo] ---------
    // VGPR r of lane l: M = r + 8*(l>=16), N = l&15 (per 16x16 f32 C layout).
    const int Ncol = lane & 15;
    float* op = out + (long)(n * COUT) * (HH * WW) + spos + Ncol;
    const int rbase = wave * 32 + ((lane >> 4) << 3);
#pragma unroll
    for (int r = 0; r < 8; ++r) {
        const long row0 = rbase + r;
        const long row1 = rbase + r + 16;
        op[row0 * (HH * WW)]      = acc00[r];   // Cout tile0, cols 0-15
        op[row0 * (HH * WW) + 16] = acc01[r];   // Cout tile0, cols 16-31
        op[row1 * (HH * WW)]      = acc10[r];   // Cout tile1, cols 0-15
        op[row1 * (HH * WW) + 16] = acc11[r];   // Cout tile1, cols 16-31
    }
}

extern "C" void kernel_launch(void* const* d_in, const int* in_sizes, int n_in,
                              void* d_out, int out_size, void* d_ws, size_t ws_size,
                              hipStream_t stream) {
    const float* x      = (const float*)d_in[0];
    const float* offset = (const float*)d_in[1];
    const float* weight = (const float*)d_in[2];
    float* out          = (float*)d_out;
    _Float16* wfrag     = (_Float16*)d_ws;   // needs 16*72*32*16*2 = 1,179,648 B

    // Pack weights into WMMA A-fragment f16 layout (589824 elements).
    pack_weight_f16<<<2304, 256, 0, stream>>>(weight, wfrag);

    // Fused deformable im2col + WMMA GEMM: 256 tiles of (32 spatial x 256 Cout).
    deform_wmma<<<NN * (HH * WW / SPW), 256, 0, stream>>>(x, offset, wfrag, out);
}
